// ReformerEncoder_35467839930468
// MI455X (gfx1250) — compile-verified, matrix operand order checked by reading, not run on hardware
//
#include <hip/hip_runtime.h>
#include <hip/hip_bf16.h>
#include <math.h>

#define USE_ASYNC_COPY 1

typedef __attribute__((ext_vector_type(16))) __bf16 v16bf;
typedef __attribute__((ext_vector_type(8)))  float  v8f;

__device__ __forceinline__ unsigned short f2bf(float f) {
  unsigned u = __builtin_bit_cast(unsigned, f);
  u = u + 0x7FFFu + ((u >> 16) & 1u);
  return (unsigned short)(u >> 16);
}

union AB16 { v16bf v; unsigned u[8]; };

constexpr int kB  = 256, kT = 52, kD = 1024, kH = 8, kDH = 128;
constexpr int kNH = 8;                 // hash rounds
constexpr int kMT = kB * kT;           // 13312 rows (104 * 128)
constexpr int kD4 = 4096;
constexpr int kNI = kNH * kT;          // 416 sorted items per head

// ---- CDNA5 async global->LDS helpers (ASYNCcnt path) ----
__device__ __forceinline__ unsigned lds_off(const void* p) {
  // generic LDS address: low 32 bits are the LDS byte offset
  return (unsigned)(unsigned long long)p;
}
__device__ __forceinline__ void async_g2l_b128(unsigned lds_addr, const void* gaddr) {
  asm volatile("global_load_async_to_lds_b128 %0, %1, off"
               :: "v"(lds_addr), "v"(gaddr) : "memory");
}
__device__ __forceinline__ void wait_async0() {
#if defined(__has_builtin) && __has_builtin(__builtin_amdgcn_s_wait_asynccnt)
  __builtin_amdgcn_s_wait_asynccnt(0);
#else
  asm volatile("s_wait_asynccnt 0" ::: "memory");
#endif
}

// ---------------- weight f32 -> bf16, WMMA-fragment-ready pack ----------------
// u32 cell = ((k>>3)*N + n)*4 + ((k>>1)&3); bf16 half = k&1.
// A k-step (32 rows of K) x 128 cols tile is then 4 contiguous 2KB segments.
__global__ void pack_w_kernel(const float* __restrict__ W,
                              unsigned short* __restrict__ out, int K, int N) {
  int e = blockIdx.x * 256 + threadIdx.x;        // K*N is a multiple of 256
  int k = e / N, n = e - k * N;
  size_t cell = ((size_t)(k >> 3) * N + n) * 4 + ((k >> 1) & 3);
  out[cell * 2 + (k & 1)] = f2bf(W[e]);
}

// ---------------- embedding + sinusoidal PE + pad mask ----------------
__global__ void embed_kernel(const int* __restrict__ x, const float* __restrict__ emb,
                             float* __restrict__ x1, float* __restrict__ x2,
                             int* __restrict__ mask) {
  int e = blockIdx.x * 256 + threadIdx.x;        // kMT*kD elements
  int row = e >> 10, d = e & 1023;
  int tok = x[row];
  int t = row % kT;
  int i2 = (d < 512) ? d : d - 512;
  float freq = expf(-(2.0f * (float)i2 / 1024.0f) * 9.210340371976184f); // ln 1e4
  float a = (float)t * freq;
  float pe = (d < 512) ? sinf(a) : cosf(a);
  float h = emb[(size_t)tok * kD + d] + pe;
  x1[e] = h; x2[e] = h;
  if (d == 0) mask[row] = (tok != 0) ? 1 : 0;
}

// ---------------- LayerNorm -> bf16 ----------------
__global__ __launch_bounds__(256) void ln_bf16_kernel(
    const float* __restrict__ X, const float* __restrict__ g,
    const float* __restrict__ b, unsigned short* __restrict__ out) {
  __shared__ float red[256];
  __shared__ float s_mean, s_inv;
  const int tid = threadIdx.x;
  const int row = blockIdx.x;
  const float* xr = X + (size_t)row * kD;
  float s = 0.f;
  for (int i = tid; i < kD; i += 256) s += xr[i];
  red[tid] = s; __syncthreads();
  for (int o = 128; o > 0; o >>= 1) { if (tid < o) red[tid] += red[tid + o]; __syncthreads(); }
  if (tid == 0) s_mean = red[0] * (1.0f / kD);
  __syncthreads();
  const float m = s_mean;
  float vv = 0.f;
  for (int i = tid; i < kD; i += 256) { float d = xr[i] - m; vv += d * d; }
  red[tid] = vv; __syncthreads();
  for (int o = 128; o > 0; o >>= 1) { if (tid < o) red[tid] += red[tid + o]; __syncthreads(); }
  if (tid == 0) s_inv = rsqrtf(red[0] * (1.0f / kD) + 1e-5f);
  __syncthreads();
  const float inv = s_inv;
  for (int i = tid; i < kD; i += 256)
    out[(size_t)row * kD + i] = f2bf((xr[i] - m) * inv * g[i] + b[i]);
}

// ---------------- bf16 WMMA GEMM: C = A(MxK) * Bpk(KxN) [+bias][+acc][gelu] ---
// EPI 0: Cf = acc ; EPI 1: Cf += acc + bias ; EPI 2: Cb = bf16(gelu(acc + bias))
template <int EPI, int N, int K>
__global__ __launch_bounds__(256) void gemm_bf16_wmma(
    const unsigned short* __restrict__ A,     // M x K bf16 row-major
    const unsigned short* __restrict__ Bpk,   // fragment-ready packed (see pack_w)
    const float* __restrict__ bias,
    float* __restrict__ Cf, unsigned short* __restrict__ Cb) {
  __shared__ unsigned sA[2][128][20];   // 128 rows x 16 u32 (32 bf16) + pad 4
  __shared__ unsigned sB[2][4 * 512];   // 4 pair-groups x 128 cols x 4 u32

  const int tid  = threadIdx.x;
  const int lane = tid & 31;
  const int wave = tid >> 5;
  const int wm   = wave & 3;            // 4 waves along M
  const int wn   = wave >> 2;           // 2 waves along N
  const int hi   = lane >> 4;
  const int l16  = lane & 15;
  const int m0   = blockIdx.y * 128;
  const int n0   = blockIdx.x * 128;

  const int arow = tid >> 1;
  const int acol = (tid & 1) * 8;                                  // u32 col
  const unsigned abase = (unsigned)(m0 + arow) * K + (tid & 1) * 16;
  const unsigned* Bu = (const unsigned*)Bpk;
  const int bg = tid >> 7;                                         // 0..1 (groups 0..3 with +2)
  const int bc = tid & 127;                                        // col in tile
  const int nk = K >> 5;

  v8f acc[2][4];
  #pragma unroll
  for (int i = 0; i < 2; ++i)
    #pragma unroll
    for (int j = 0; j < 4; ++j)
      #pragma unroll
      for (int r = 0; r < 8; ++r) acc[i][j][r] = 0.0f;

#if USE_ASYNC_COPY
  const unsigned sAoff  = lds_off(&sA[0][arow][acol]);
  const unsigned sBoff0 = lds_off(&sB[0][(bg)     * 512 + bc * 4]);
  const unsigned sBoff1 = lds_off(&sB[0][(bg + 2) * 512 + bc * 4]);
  auto issue_tile = [&](int buf, int ks) {
    const unsigned ao = abase + (unsigned)ks * 32;
    async_g2l_b128(sAoff + (unsigned)buf * 10240u,        A + ao);
    async_g2l_b128(sAoff + (unsigned)buf * 10240u + 16u,  A + ao + 8);
    async_g2l_b128(sBoff0 + (unsigned)buf * 8192u,
                   Bu + ((unsigned)(ks * 4 + bg) * N + n0 + bc) * 4);
    async_g2l_b128(sBoff1 + (unsigned)buf * 8192u,
                   Bu + ((unsigned)(ks * 4 + bg + 2) * N + n0 + bc) * 4);
  };
  issue_tile(0, 0);
  wait_async0();
  __syncthreads();
#else
  uint4 a0 = *(const uint4*)(A + abase);
  uint4 a1 = *(const uint4*)(A + abase + 8);
  uint4 b0 = *(const uint4*)(Bu + ((unsigned)(bg) * N + n0 + bc) * 4);
  uint4 b1 = *(const uint4*)(Bu + ((unsigned)(bg + 2) * N + n0 + bc) * 4);
  *(uint4*)&sA[0][arow][acol]          = a0;
  *(uint4*)&sA[0][arow][acol + 4]      = a1;
  *(uint4*)&sB[0][(bg) * 512 + bc * 4]     = b0;
  *(uint4*)&sB[0][(bg + 2) * 512 + bc * 4] = b1;
  __syncthreads();
#endif

  for (int ks = 0; ks < nk; ++ks) {
    const int cur = ks & 1;
    const bool more = (ks + 1) < nk;
    if (more) {
#if USE_ASYNC_COPY
      issue_tile(cur ^ 1, ks + 1);
#else
      const unsigned ao = abase + (unsigned)(ks + 1) * 32;
      a0 = *(const uint4*)(A + ao);
      a1 = *(const uint4*)(A + ao + 8);
      b0 = *(const uint4*)(Bu + ((unsigned)((ks + 1) * 4 + bg) * N + n0 + bc) * 4);
      b1 = *(const uint4*)(Bu + ((unsigned)((ks + 1) * 4 + bg + 2) * N + n0 + bc) * 4);
#endif
      if (ks + 2 < nk) {
        __builtin_prefetch(A + abase + (unsigned)(ks + 2) * 32, 0, 0);
        __builtin_prefetch(Bu + ((unsigned)((ks + 2) * 4 + bg) * N + n0 + bc) * 4, 0, 0);
      }
    }

    AB16 af[2], bfr[4];
    #pragma unroll
    for (int mt = 0; mt < 2; ++mt) {
      const int row = wm * 32 + mt * 16 + l16;
      *(uint4*)&af[mt].u[0] = *(const uint4*)&sA[cur][row][hi * 4];
      *(uint4*)&af[mt].u[4] = *(const uint4*)&sA[cur][row][8 + hi * 4];
    }
    #pragma unroll
    for (int nt = 0; nt < 4; ++nt) {
      const int col = wn * 64 + nt * 16 + l16;
      *(uint4*)&bfr[nt].u[0] = *(const uint4*)&sB[cur][(hi)     * 512 + col * 4];
      *(uint4*)&bfr[nt].u[4] = *(const uint4*)&sB[cur][(2 + hi) * 512 + col * 4];
    }
    #pragma unroll
    for (int mt = 0; mt < 2; ++mt)
      #pragma unroll
      for (int nt = 0; nt < 4; ++nt)
        acc[mt][nt] = __builtin_amdgcn_wmma_f32_16x16x32_bf16(
            false, af[mt].v, false, bfr[nt].v, (short)0, acc[mt][nt], false, false);

    if (more) {
#if USE_ASYNC_COPY
      wait_async0();
      __syncthreads();
#else
      __syncthreads();
      const int nxt = cur ^ 1;
      *(uint4*)&sA[nxt][arow][acol]          = a0;
      *(uint4*)&sA[nxt][arow][acol + 4]      = a1;
      *(uint4*)&sB[nxt][(bg) * 512 + bc * 4]     = b0;
      *(uint4*)&sB[nxt][(bg + 2) * 512 + bc * 4] = b1;
      __syncthreads();
#endif
    }
  }

  // epilogue: one base per 16x16 tile, r*N folds into immediate offsets
  #pragma unroll
  for (int mt = 0; mt < 2; ++mt)
    #pragma unroll
    for (int nt = 0; nt < 4; ++nt) {
      const int gm0 = m0 + wm * 32 + mt * 16 + hi * 8;
      const int gn  = n0 + wn * 64 + nt * 16 + l16;
      const unsigned idx0 = (unsigned)gm0 * N + gn;
      #pragma unroll
      for (int r = 0; r < 8; ++r) {
        float a = acc[mt][nt][r];
        if constexpr (EPI == 0) {
          Cf[idx0 + r * N] = a;
        } else if constexpr (EPI == 1) {
          Cf[idx0 + r * N] = Cf[idx0 + r * N] + a + bias[gn];
        } else {
          float xg = a + bias[gn];
          float ge = 0.5f * xg * (1.0f + erff(xg * 0.70710678118654752f));
          Cb[idx0 + r * N] = f2bf(ge);
        }
      }
    }
}

// ---------------- LSH attention: one block per (batch, head) ----------------
__global__ __launch_bounds__(256) void attn_kernel(
    const float* __restrict__ qk, const float* __restrict__ v,
    const int* __restrict__ mask, const float* __restrict__ R,
    unsigned short* __restrict__ obf) {
  constexpr int QS = 132;                          // padded stride (banks)
  __shared__ float s_qk[kT * QS];
  __shared__ float s_out[kT * kDH];
  __shared__ float s_invn[kT];
  __shared__ int   s_m[kT];
  __shared__ int   s_key[kNI];
  __shared__ int   s_st[kNI];
  __shared__ int   s_pos[kNI];
  __shared__ float s_lse[kNI];
  __shared__ float s_d1[kNI];
  __shared__ float s_w[kNI];

  const int tid = threadIdx.x;
  const int bh = blockIdx.x;
  const int b = bh >> 3;
  const int h = bh & 7;
  const size_t rowbase = ((size_t)b * kT) * kD + (size_t)h * kDH;

  for (int e = tid; e < kT * kDH; e += 256) {
    int t = e >> 7, f = e & 127;
    s_qk[t * QS + f] = qk[rowbase + (size_t)t * kD + f];
  }
  __syncthreads();

  if (tid < kT) {
    float s = 0.f;
    for (int f = 0; f < kDH; ++f) { float q = s_qk[tid * QS + f]; s += q * q; }
    s_invn[tid] = 1.0f / fmaxf(sqrtf(s), 1e-12f);
    s_m[tid] = mask[b * kT + tid];
  }
  __syncthreads();

  // rotations + argmax over [rot, -rot] -> bucket key
  for (int it = tid; it < kNI; it += 256) {
    int r = it / kT, t = it - r * kT;
    float dv[26];
    #pragma unroll
    for (int i = 0; i < 26; ++i) dv[i] = 0.f;
    for (int f = 0; f < kDH; ++f) {
      float q = s_qk[t * QS + f];
      const float* rp = R + ((size_t)f * kNH + r) * 26;
      #pragma unroll
      for (int i = 0; i < 26; ++i) dv[i] += q * rp[i];
    }
    float best = -3.4e38f; int bkt = 0;
    #pragma unroll
    for (int i = 0; i < 26; ++i) if (dv[i] > best) { best = dv[i]; bkt = i; }
    #pragma unroll
    for (int i = 0; i < 26; ++i) if (-dv[i] > best) { best = -dv[i]; bkt = 26 + i; }
    s_key[it] = bkt * kT + t;
  }
  __syncthreads();

  // stable rank sort within each hash round (keys distinct: bucket*T + t)
  for (int it = tid; it < kNI; it += 256) {
    int r = it / kT, t = it - r * kT;
    int k = s_key[it];
    int rank = 0;
    for (int j = 0; j < kT; ++j) rank += (s_key[r * kT + j] < k) ? 1 : 0;
    s_st[r * kT + rank] = t;
    s_pos[it] = r * kT + rank;
  }
  __syncthreads();

  // chunked attention, chunk=1 with look-one-back over the global sorted ring
  for (int i = tid; i < kNI; i += 256) {
    int t  = s_st[i];
    int ip = (i + kNI - 1) % kNI;
    int tp = s_st[ip];
    float dot = 0.f;
    for (int f = 0; f < kDH; ++f) dot += s_qk[t * QS + f] * s_qk[tp * QS + f];
    float d1 = dot * s_invn[tp] * 0.0883883476483184f;   // DH^-0.5
    if (!(s_m[t] && s_m[tp])) d1 = -1e9f;
    if (t == tp) d1 = -5e4f;
    const float d0 = -5e4f;                              // self token always
    float mx = fmaxf(d0, d1);
    float l = mx + logf(expf(d0 - mx) + expf(d1 - mx));
    s_lse[i] = l;
    s_d1[i]  = d1;
  }
  __syncthreads();

  // combine hash rounds: softmax of lse over rounds per token
  if (tid < kT) {
    float lo[kNH];
    float mx = -3.4e38f;
    for (int r = 0; r < kNH; ++r) { lo[r] = s_lse[s_pos[r * kT + tid]]; mx = fmaxf(mx, lo[r]); }
    float sum = 0.f;
    for (int r = 0; r < kNH; ++r) { lo[r] = expf(lo[r] - mx); sum += lo[r]; }
    float inv = 1.0f / sum;
    for (int r = 0; r < kNH; ++r) s_w[r * kT + tid] = lo[r] * inv;
  }
  for (int e = tid; e < kT * kDH; e += 256) s_out[e] = 0.f;
  __syncthreads();

  for (int r = 0; r < kNH; ++r) {
    for (int e = tid; e < kT * kDH; e += 256) {
      int il = e >> 7, f = e & 127;
      int i  = r * kT + il;
      int t  = s_st[i];
      int tp = s_st[(i + kNI - 1) % kNI];
      float lse = s_lse[i];
      float e0 = expf(-5e4f - lse);
      float e1 = expf(s_d1[i] - lse);
      float vt = v[rowbase + (size_t)t  * kD + f];
      float vp = v[rowbase + (size_t)tp * kD + f];
      s_out[t * kDH + f] += s_w[r * kT + t] * (e0 * vt + e1 * vp);
    }
    __syncthreads();
  }

  for (int e = tid; e < kT * kDH; e += 256) {
    int t = e >> 7, f = e & 127;
    obf[rowbase + (size_t)t * kD + f] = f2bf(s_out[t * kDH + f]);
  }
}

// ---------------- final: out[b,:] = 0.5*(x1[b,0,:] + x2[b,0,:]) ----------------
__global__ void final_kernel(const float* __restrict__ x1, const float* __restrict__ x2,
                             float* __restrict__ out) {
  int e = blockIdx.x * 256 + threadIdx.x;      // kB*kD
  int b = e >> 10, d = e & 1023;
  size_t idx = (size_t)b * kT * kD + d;
  out[e] = 0.5f * (x1[idx] + x2[idx]);
}

extern "C" void kernel_launch(void* const* d_in, const int* in_sizes, int n_in,
                              void* d_out, int out_size, void* d_ws, size_t ws_size,
                              hipStream_t stream) {
  (void)in_sizes; (void)n_in; (void)out_size; (void)ws_size;
  const int*   x    = (const int*)  d_in[0];
  const float* emb  = (const float*)d_in[1];
  const float* ln1g = (const float*)d_in[2];
  const float* ln1b = (const float*)d_in[3];
  const float* Wqk  = (const float*)d_in[4];
  const float* Wv   = (const float*)d_in[5];
  const float* Wo   = (const float*)d_in[6];
  const float* bo   = (const float*)d_in[7];
  const float* ln2g = (const float*)d_in[8];
  const float* ln2b = (const float*)d_in[9];
  const float* W1   = (const float*)d_in[10];
  const float* b1   = (const float*)d_in[11];
  const float* W2   = (const float*)d_in[12];
  const float* b2   = (const float*)d_in[13];
  const float* rot  = (const float*)d_in[14];

  char* p = (char*)d_ws;
  auto carve = [&](size_t bytes) { char* q = p; p += (bytes + 255) & ~(size_t)255; return q; };

  unsigned short *wqk_bf[2], *wv_bf[2], *wo_bf[2], *w1_bf[2], *w2_bf[2];
  for (int l = 0; l < 2; ++l) {
    wqk_bf[l] = (unsigned short*)carve((size_t)kD * kD * 2);
    wv_bf[l]  = (unsigned short*)carve((size_t)kD * kD * 2);
    wo_bf[l]  = (unsigned short*)carve((size_t)kD * kD * 2);
    w1_bf[l]  = (unsigned short*)carve((size_t)kD * kD4 * 2);
    w2_bf[l]  = (unsigned short*)carve((size_t)kD4 * kD * 2);
  }
  float* x1 = (float*)carve((size_t)kMT * kD * 4);
  float* x2 = (float*)carve((size_t)kMT * kD * 4);
  unsigned short* xn  = (unsigned short*)carve((size_t)kMT * kD * 2);
  unsigned short* obf = (unsigned short*)carve((size_t)kMT * kD * 2);
  char* bigA = carve((size_t)kMT * kD4 * 4);     // qk|v (f32) OR FFN mid (bf16)
  float* qkbuf = (float*)bigA;
  float* vbuf  = (float*)(bigA + (size_t)kMT * kD * 4);
  unsigned short* mid = (unsigned short*)bigA;
  int* maskbuf = (int*)carve((size_t)kMT * 4);

  dim3 blk(256);
  for (int l = 0; l < 2; ++l) {
    pack_w_kernel<<<(kD * kD) / 256,  blk, 0, stream>>>(Wqk + (size_t)l * kD * kD,  wqk_bf[l], kD, kD);
    pack_w_kernel<<<(kD * kD) / 256,  blk, 0, stream>>>(Wv  + (size_t)l * kD * kD,  wv_bf[l],  kD, kD);
    pack_w_kernel<<<(kD * kD) / 256,  blk, 0, stream>>>(Wo  + (size_t)l * kD * kD,  wo_bf[l],  kD, kD);
    pack_w_kernel<<<(kD * kD4) / 256, blk, 0, stream>>>(W1  + (size_t)l * kD * kD4, w1_bf[l],  kD, kD4);
    pack_w_kernel<<<(kD4 * kD) / 256, blk, 0, stream>>>(W2  + (size_t)l * kD4 * kD, w2_bf[l],  kD4, kD);
  }
  embed_kernel<<<(kMT * kD) / 256, blk, 0, stream>>>(x, emb, x1, x2, maskbuf);

  for (int l = 0; l < 2; ++l) {
    ln_bf16_kernel<<<kMT, blk, 0, stream>>>(x2, ln1g + l * kD, ln1b + l * kD, xn);
    gemm_bf16_wmma<0, kD, kD><<<dim3(kD / 128, kMT / 128), blk, 0, stream>>>(
        xn, wqk_bf[l], nullptr, qkbuf, nullptr);
    gemm_bf16_wmma<0, kD, kD><<<dim3(kD / 128, kMT / 128), blk, 0, stream>>>(
        xn, wv_bf[l], nullptr, vbuf, nullptr);
    attn_kernel<<<kB * kH, blk, 0, stream>>>(
        qkbuf, vbuf, maskbuf, rot + (size_t)l * kDH * kNH * 26, obf);
    gemm_bf16_wmma<1, kD, kD><<<dim3(kD / 128, kMT / 128), blk, 0, stream>>>(
        obf, wo_bf[l], bo + l * kD, x1, nullptr);
    ln_bf16_kernel<<<kMT, blk, 0, stream>>>(x1, ln2g + l * kD, ln2b + l * kD, xn);
    gemm_bf16_wmma<2, kD4, kD><<<dim3(kD4 / 128, kMT / 128), blk, 0, stream>>>(
        xn, w1_bf[l], b1 + l * kD4, nullptr, mid);
    gemm_bf16_wmma<1, kD, kD4><<<dim3(kD / 128, kMT / 128), blk, 0, stream>>>(
        mid, w2_bf[l], b2 + l * kD, x2, nullptr);
  }
  final_kernel<<<(kB * kD) / 256, blk, 0, stream>>>(x1, x2, (float*)d_out);
}
